// DenseNet_9852654977027
// MI455X (gfx1250) — 8-bit matrix operand order repaired and checked by reading, compiled, not run on hardware
//
#include <hip/hip_runtime.h>
#include <hip/hip_bf16.h>
#include <cstdint>

typedef __attribute__((ext_vector_type(8))) int v8i;
typedef int v4i_vs __attribute__((__vector_size__(4 * sizeof(int))));

#define EPSF 1e-5f

#if __has_builtin(__builtin_amdgcn_global_load_async_to_lds_b128) && \
    __has_builtin(__builtin_amdgcn_s_wait_asynccnt)
#define USE_ASYNC_B 1
#else
#define USE_ASYNC_B 0
#endif

__global__ void zero_kernel(float* __restrict__ p, int n) {
  int i = blockIdx.x * blockDim.x + threadIdx.x;
  if (i < n) p[i] = 0.0f;
}

__global__ void repack_kernel(const float* __restrict__ w, int8_t* __restrict__ wb,
                              int Cin, int CB, int Kpad, int total) {
  int idx = blockIdx.x * blockDim.x + threadIdx.x;
  if (idx >= total) return;
  int k  = idx % Kpad;
  int co = idx / Kpad;
  int j  = k >> 6;
  int t  = k & 63;
  int khw = j / CB;
  int cb  = j - khw * CB;
  int ci  = (cb << 6) + t;
  int8_t v = 0;
  if (ci < Cin) {
    float wv = w[((size_t)co * Cin + ci) * 9 + khw];
    v = (wv >= 0.0f) ? (int8_t)1 : (int8_t)-1;
  }
  wb[idx] = v;
}

__global__ void conv1_kernel(const float* __restrict__ x, const float* __restrict__ w,
                             float* __restrict__ raw, int total) {
  int idx = blockIdx.x * blockDim.x + threadIdx.x;
  if (idx >= total) return;
  int co = idx & 127;
  int m  = idx >> 7;
  int n  = m / (126 * 126);
  int rem = m - n * (126 * 126);
  int oy = rem / 126;
  int ox = rem - oy * 126;
  const float* xp = x + ((size_t)n * 3 + 1) * (128 * 128);
  float s = 0.0f;
#pragma unroll
  for (int kh = 0; kh < 3; ++kh) {
#pragma unroll
    for (int kw = 0; kw < 3; ++kw) {
      float wv  = w[co * 9 + kh * 3 + kw];
      float sgn = (wv >= 0.0f) ? 1.0f : -1.0f;
      s += sgn * xp[(oy + kh) * 128 + (ox + kw)];
    }
  }
  raw[idx] = s;
}

__global__ __launch_bounds__(32) void bconv_wmma_kernel(
    const int8_t* __restrict__ act, int Batch, int Hi, int Wi, int Cin, int CinStride,
    const int8_t* __restrict__ wb, int Cout, int CB, int Kpad,
    int strd, int pad, int Ho, int Wo, float* __restrict__ raw) {
  __shared__ int As4[1024];
  __shared__ int Bs4[512];
  __shared__ int rNHi[64];
  __shared__ int rOys[64];
  __shared__ int rOxs[64];

  const int lane  = threadIdx.x;
  const int mtile = blockIdx.x * 64;
  const int ctile = blockIdx.y * 32;
  const int HW    = Ho * Wo;
  const int Mtot  = Batch * HW;
  const int hi    = lane >> 4;
  const int ml    = lane & 15;

  for (int r = lane; r < 64; r += 32) {
    int Midx = mtile + r;
    int nhi = 0, oys = -0x40000000, oxs = -0x40000000;
    if (Midx < Mtot) {
      int n   = Midx / HW;
      int rem = Midx - n * HW;
      int oy  = rem / Wo;
      int ox  = rem - oy * Wo;
      nhi = n * Hi;
      oys = oy * strd - pad;
      oxs = ox * strd - pad;
    }
    rNHi[r] = nhi;
    rOys[r] = oys;
    rOxs[r] = oxs;
  }
  __syncthreads();

  v8i acc[4][2];
#pragma unroll
  for (int mt = 0; mt < 4; ++mt)
#pragma unroll
    for (int nt = 0; nt < 2; ++nt) acc[mt][nt] = (v8i){};

  const int nchunks = 9 * CB;
  int chunk = 0;
  for (int khw = 0; khw < 9; ++khw) {
    const int kh = khw / 3;
    const int kw = khw - kh * 3;
    for (int cb = 0; cb < CB; ++cb, ++chunk) {
      const int ci0 = cb << 6;

#if USE_ASYNC_B
      {
        int colb = lane >> 2;
        int boff = (lane & 3) << 4;
#pragma unroll
        for (int q = 0; q < 4; ++q) {
          const int8_t* gsrc = wb + (size_t)(ctile + q * 8 + colb) * Kpad +
                               (chunk << 6) + boff;
          __builtin_amdgcn_global_load_async_to_lds_b128(
              (v4i_vs __attribute__((address_space(1)))*)(gsrc),
              (v4i_vs __attribute__((address_space(3)))*)((char*)Bs4 + q * 512 + lane * 16),
              0, 0);
        }
      }
#else
      for (int i = lane; i < 512; i += 32) {
        int col = i >> 4;
        Bs4[i] = *(const int*)(wb + (size_t)(ctile + col) * Kpad + (chunk << 6) + ((i & 15) << 2));
      }
#endif

      for (int i = lane; i < 1024; i += 32) {
        int row = i >> 4;
        int ci  = ci0 + ((i & 15) << 2);
        int iy  = rOys[row] + kh;
        int ix  = rOxs[row] + kw;
        int v   = 0;
        if (((unsigned)iy < (unsigned)Hi) & ((unsigned)ix < (unsigned)Wi) & (ci < Cin)) {
          const int8_t* p = act + (size_t)((rNHi[row] + iy) * Wi + ix) * (size_t)CinStride + ci;
          v = *(const int*)p;
        }
        As4[i] = v;
      }

      if (chunk + 1 < nchunks)
        __builtin_prefetch(wb + (size_t)(ctile + lane) * Kpad + ((chunk + 1) << 6), 0, 1);

#if USE_ASYNC_B
      __builtin_amdgcn_s_wait_asynccnt(0);
#endif
      __syncthreads();

      v8i bf[2];
#pragma unroll
      for (int nt = 0; nt < 2; ++nt) {
#pragma unroll
        for (int v = 0; v < 8; ++v) {
          int kb = (hi << 4) + ((v & 3) << 2) + ((v & 4) << 3);
          bf[nt][v] = Bs4[(nt * 16 + ml) * 16 + (kb >> 2)];
        }
      }
#pragma unroll
      for (int mt = 0; mt < 4; ++mt) {
        v8i af;
#pragma unroll
        for (int v = 0; v < 8; ++v) {
          int vv = v & 3;
          int ka = (hi << 4) + ((vv) << 2) + ((v & 4) << 3);
          af[v] = As4[(mt * 16 + ml) * 16 + (ka >> 2)];
        }
        acc[mt][0] = __builtin_amdgcn_wmma_i32_16x16x64_iu8(true, af, true, bf[0],
                                                            acc[mt][0], false, false);
        acc[mt][1] = __builtin_amdgcn_wmma_i32_16x16x64_iu8(true, af, true, bf[1],
                                                            acc[mt][1], false, false);
      }
      __syncthreads();
    }
  }

#pragma unroll
  for (int mt = 0; mt < 4; ++mt) {
#pragma unroll
    for (int nt = 0; nt < 2; ++nt) {
      int col = ctile + nt * 16 + ml;
#pragma unroll
      for (int r = 0; r < 8; ++r) {
        int m = mtile + mt * 16 + r + (hi << 3);
        if (m < Mtot) raw[(size_t)m * Cout + col] = (float)acc[mt][nt][r];
      }
    }
  }
}

__global__ void bn_stats_kernel(const float* __restrict__ raw, int M, int C,
                                float* __restrict__ stats) {
  int c = blockIdx.x * blockDim.x + threadIdx.x;
  if (c >= C) return;
  float s = 0.0f, s2 = 0.0f;
  for (int m = blockIdx.y; m < M; m += gridDim.y) {
    float v = raw[(size_t)m * C + c];
    s += v;
    s2 += v * v;
  }
  atomicAdd(&stats[c], s);
  atomicAdd(&stats[C + c], s2);
}

__global__ void bn_sign_kernel(const float* __restrict__ raw, const float* __restrict__ stats,
                               const float* __restrict__ g, const float* __restrict__ b,
                               int M, int C, int8_t* __restrict__ dst,
                               int dstStride, int chanOff, int total) {
  int idx = blockIdx.x * blockDim.x + threadIdx.x;
  if (idx >= total) return;
  int c = idx % C;
  int m = idx / C;
  float invM  = 1.0f / (float)M;
  float mean  = stats[c] * invM;
  float var   = stats[C + c] * invM - mean * mean;
  float xh    = (raw[idx] - mean) * __frsqrt_rn(var + EPSF);
  float y     = xh * g[c] + b[c];
  dst[(size_t)m * dstStride + chanOff + c] = (y >= 0.0f) ? (int8_t)1 : (int8_t)-1;
}

__global__ void fc_kernel(const int8_t* __restrict__ act, const float* __restrict__ w,
                          const float* __restrict__ bias, float* __restrict__ pre) {
  const int F = 1472 * 256;
  int n   = blockIdx.x / 3;
  int cls = blockIdx.x % 3;
  int t   = threadIdx.x;
  int s = 0;
  for (int j = t; j < F; j += 256) {
    int c  = j >> 8;
    int hw = j & 255;
    int a  = (int)act[((size_t)n * 256 + hw) * 1472 + c];
    float wv = w[(size_t)cls * F + j];
    s += (wv >= 0.0f) ? a : -a;
  }
  __shared__ int red[256];
  red[t] = s;
  __syncthreads();
  for (int off = 128; off > 0; off >>= 1) {
    if (t < off) red[t] += red[t + off];
    __syncthreads();
  }
  if (t == 0) pre[n * 3 + cls] = (float)red[0] + bias[cls];
}

__global__ void bn1d_kernel(const float* __restrict__ pre, const float* __restrict__ g,
                            const float* __restrict__ bb, float* __restrict__ out) {
  int c = threadIdx.x;
  if (c >= 3) return;
  float m = 0.0f;
  for (int n = 0; n < 8; ++n) m += pre[n * 3 + c];
  m *= 0.125f;
  float v = 0.0f;
  for (int n = 0; n < 8; ++n) {
    float d = pre[n * 3 + c] - m;
    v += d * d;
  }
  v *= 0.125f;
  float inv = __frsqrt_rn(v + EPSF);
  for (int n = 0; n < 8; ++n)
    out[n * 3 + c] = (pre[n * 3 + c] - m) * inv * g[c] + bb[c];
}

extern "C" void kernel_launch(void* const* d_in, const int* in_sizes, int n_in,
                              void* d_out, int out_size, void* d_ws, size_t ws_size,
                              hipStream_t stream) {
  (void)in_sizes; (void)n_in; (void)out_size; (void)ws_size;

  struct BGW { int b, g, w; };
  int p = 0;
  auto take3 = [&]() { BGW r; r.b = p++; r.g = p++; r.w = p++; return r; };
  BGW c1 = take3();
  BGW d1[6];  for (int i = 0; i < 6;  ++i) d1[i] = take3();
  BGW d2[12]; for (int i = 0; i < 12; ++i) d2[i] = take3();
  BGW d3[24]; for (int i = 0; i < 24; ++i) d3[i] = take3();
  int i_fc_b = p++, i_fc_bb = p++, i_fc_g = p++, i_fc_w = p++;
  BGW t1 = take3(), t2 = take3(), t3 = take3();
  int i_x = p++;

  auto F = [&](int i) { return (const float*)d_in[i]; };

  char* ws = (char*)d_ws;
  size_t off = 0;
  auto alloc = [&](size_t bytes) {
    size_t o = off;
    off = (off + bytes + 255) & ~(size_t)255;
    return o;
  };
  const int Nb = 8;
  const int M1 = Nb * 126 * 126;
  const int M2 = Nb * 63 * 63;
  const int M3 = Nb * 32 * 32;
  const int M4 = Nb * 16 * 16;

  int8_t* A1  = (int8_t*)(ws + alloc((size_t)M1 * 320));
  int8_t* A2  = (int8_t*)(ws + alloc((size_t)M2 * 704));
  int8_t* A3  = (int8_t*)(ws + alloc((size_t)M3 * 1472));
  int8_t* A4  = (int8_t*)(ws + alloc((size_t)M4 * 1472));
  float*  RAW = (float*)(ws + alloc((size_t)M1 * 128 * 4));
  float*  STATS = (float*)(ws + alloc((size_t)4096 * 4));
  float*  PRE   = (float*)(ws + alloc(256));
  int8_t* WSC   = (int8_t*)(ws + alloc((size_t)1472 * 13248));

  auto zero = [&](float* ptr, int n) {
    zero_kernel<<<(n + 255) / 256, 256, 0, stream>>>(ptr, n);
  };

  auto run_bconv = [&](const int8_t* actIn, int Hi, int Wi, int Cin, int CinStride,
                       BGW prm, int Cout, int strd, int pad, int Ho, int Wo,
                       int8_t* actOut, int outStride, int chanOff) {
    int CB    = (Cin + 63) >> 6;
    int Kpad  = 9 * CB * 64;
    int tw    = Cout * Kpad;
    repack_kernel<<<(tw + 255) / 256, 256, 0, stream>>>(F(prm.w), WSC, Cin, CB, Kpad, tw);
    int M = Nb * Ho * Wo;
    dim3 grid((M + 63) / 64, Cout / 32);
    bconv_wmma_kernel<<<grid, 32, 0, stream>>>(actIn, Nb, Hi, Wi, Cin, CinStride,
                                               WSC, Cout, CB, Kpad,
                                               strd, pad, Ho, Wo, RAW);
    zero(STATS, 2 * Cout);
    bn_stats_kernel<<<dim3((Cout + 255) / 256, 64), 256, 0, stream>>>(RAW, M, Cout, STATS);
    int tot = M * Cout;
    bn_sign_kernel<<<(tot + 255) / 256, 256, 0, stream>>>(RAW, STATS, F(prm.g), F(prm.b),
                                                          M, Cout, actOut, outStride,
                                                          chanOff, tot);
  };

  {
    int tot = M1 * 128;
    conv1_kernel<<<(tot + 255) / 256, 256, 0, stream>>>(F(i_x), F(c1.w), RAW, tot);
    zero(STATS, 2 * 128);
    bn_stats_kernel<<<dim3(1, 64), 256, 0, stream>>>(RAW, M1, 128, STATS);
    bn_sign_kernel<<<(tot + 255) / 256, 256, 0, stream>>>(RAW, STATS, F(c1.g), F(c1.b),
                                                          M1, 128, A1, 320, 0, tot);
  }

  int np = 128;
  for (int i = 0; i < 6; ++i) {
    run_bconv(A1, 126, 126, np, 320, d1[i], 32, 1, 1, 126, 126, A1, 320, np);
    np += 32;
  }
  run_bconv(A1, 126, 126, 320, 320, t1, 320, 2, 1, 63, 63, A2, 704, 0);

  np = 320;
  for (int i = 0; i < 12; ++i) {
    run_bconv(A2, 63, 63, np, 704, d2[i], 32, 1, 1, 63, 63, A2, 704, np);
    np += 32;
  }
  run_bconv(A2, 63, 63, 704, 704, t2, 704, 2, 1, 32, 32, A3, 1472, 0);

  np = 704;
  for (int i = 0; i < 24; ++i) {
    run_bconv(A3, 32, 32, np, 1472, d3[i], 32, 1, 1, 32, 32, A3, 1472, np);
    np += 32;
  }
  run_bconv(A3, 32, 32, 1472, 1472, t3, 1472, 2, 1, 16, 16, A4, 1472, 0);

  fc_kernel<<<24, 256, 0, stream>>>(A4, F(i_fc_w), F(i_fc_b), PRE);
  bn1d_kernel<<<1, 32, 0, stream>>>(PRE, F(i_fc_g), F(i_fc_bb), (float*)d_out);
}
